// PEER_20693152432472
// MI455X (gfx1250) — matrix operand order repaired and checked by reading, not compiled, not run on hardware
//
#include <hip/hip_runtime.h>
#include <math.h>

typedef __attribute__((ext_vector_type(16))) _Float16 v16h;
typedef __attribute__((ext_vector_type(8)))  float    v8f;

#define T_TOK 2048   // b*s tokens
#define NDIM  1024
#define NFEAT 2048   // 2*H*KD
#define NH    8
#define KDIM  128
#define NKEY  256
#define TKP   8

// ---------------------------------------------------------------- helpers
__device__ __forceinline__ float gelu_new_f(float x) {
    const float c = 0.7978845608028654f; // sqrt(2/pi)
    return 0.5f * x * (1.0f + tanhf(c * (x + 0.044715f * x * x * x)));
}

__device__ __forceinline__ void top8_insert(float (&bv)[8], int (&bi)[8], float v, int idx) {
    if (v <= bv[7]) return;
    int pos = 7;
    while (pos > 0 && v > bv[pos - 1]) { bv[pos] = bv[pos - 1]; bi[pos] = bi[pos - 1]; --pos; }
    bv[pos] = v; bi[pos] = idx;
}

// ---------------------------------------------------------------- prep: x -> f16
__global__ void k_cvt_x(const float* __restrict__ x, _Float16* __restrict__ xh, int n) {
    int i = blockIdx.x * 256 + threadIdx.x;
    if (i < n) xh[i] = (_Float16)x[i];
}

// w_q [1024 x 2048] -> WMMA B-fragment layout:
// linear id = ((ntile*32 + kstep)*32 + lane)*16 + i ; element = B[kstep*32 + (lane&16) + i][ntile*16 + (lane&15)]
__global__ void k_cvt_wfrag(const float* __restrict__ w, _Float16* __restrict__ wf) {
    int id = blockIdx.x * 256 + threadIdx.x;          // 2^21 total
    int i     = id & 15;
    int lane  = (id >> 4) & 31;
    int kstep = (id >> 9) & 31;
    int ntile = id >> 14;                             // 0..127
    int col = ntile * 16 + (lane & 15);
    int k   = kstep * 32 + (lane & 16) + i;
    wf[id] = (_Float16)w[k * NFEAT + col];
}

// keys [H][256][2][128] -> fragment layout per (p,h,ntile,kstep): B[k][key]
__global__ void k_cvt_keysf(const float* __restrict__ keys, _Float16* __restrict__ kf) {
    int id = blockIdx.x * 256 + threadIdx.x;          // 2^19 total
    int i     = id & 15;
    int lane  = (id >> 4) & 31;
    int kstep = (id >> 9) & 3;
    int ntile = (id >> 11) & 15;
    int h     = (id >> 15) & 7;
    int p     = (id >> 18) & 1;
    int col = ntile * 16 + (lane & 15);
    int k   = kstep * 32 + (lane & 16) + i;
    kf[id] = (_Float16)keys[((h * NKEY + col) * 2 + p) * KDIM + k];
}

// ---------------------------------------------------------------- q = x @ w_q (WMMA f16->f32)
__global__ __launch_bounds__(256) void k_gemm_q(const _Float16* __restrict__ xh,
                                                const _Float16* __restrict__ wf,
                                                float* __restrict__ q) {
    __shared__ _Float16 As[16 * 1024];                // full K row-block, 32KB
    int tid = threadIdx.x, lane = tid & 31, wave = tid >> 5;
    int m0 = blockIdx.y * 16;
    int ntile = blockIdx.x * 8 + wave;

    { // stage A (coalesced dword copies)
        const unsigned int* src = (const unsigned int*)xh + m0 * 512;
        unsigned int* dst = (unsigned int*)As;
        for (int li = tid; li < 8192; li += 256) {
            int row = li >> 9, cu = li & 511;
            dst[li] = src[row * 512 + cu];
        }
    }
    __syncthreads();

    v8f c;
    #pragma unroll
    for (int r = 0; r < 8; ++r) c[r] = 0.0f;

    int rowA  = lane & 15;
    int kOffA = (lane & 16);                          // 0 or 16
    const v16h* bptr = (const v16h*)wf + (ntile * 32) * 32 + lane;
    for (int ks = 0; ks < 32; ++ks) {
        v16h a;
        const _Float16* ar = As + rowA * 1024 + ks * 32;
        #pragma unroll
        for (int i = 0; i < 8; ++i) a[i] = ar[kOffA + i];
        #pragma unroll
        for (int i = 0; i < 8; ++i) a[8 + i] = ar[kOffA + 8 + i];
        v16h b = bptr[ks * 32];
        c = __builtin_amdgcn_wmma_f32_16x16x32_f16(false, a, false, b, (short)0, c, false, false);
    }
    int rBase = m0 + ((lane & 16) >> 1);
    int cCol  = ntile * 16 + (lane & 15);
    #pragma unroll
    for (int r = 0; r < 8; ++r)
        q[(rBase + r) * NFEAT + cCol] = c[r];
}

// ---------------------------------------------------------------- BatchNorm batch stats -> scale/shift
__global__ void k_bn(const float* __restrict__ q, const float* __restrict__ gamma,
                     const float* __restrict__ beta, float* __restrict__ scale,
                     float* __restrict__ shift) {
    int f = blockIdx.x * 256 + threadIdx.x;           // 2048 features
    float s = 0.f, s2 = 0.f;
    for (int t = 0; t < T_TOK; ++t) {
        float v = q[t * NFEAT + f];
        s += v; s2 += v * v;
    }
    float mean = s * (1.0f / T_TOK);
    float var  = s2 * (1.0f / T_TOK) - mean * mean;   // biased var (matches jnp.var)
    float rstd = rsqrtf(var + 1e-5f);
    float sc = gamma[f] * rstd;
    scale[f] = sc;
    shift[f] = beta[f] - mean * sc;
}

__global__ void k_qn(const float* __restrict__ q, const float* __restrict__ scale,
                     const float* __restrict__ shift, _Float16* __restrict__ qn) {
    int i = blockIdx.x * 256 + threadIdx.x;           // 4M elements
    int f = i & (NFEAT - 1);
    qn[i] = (_Float16)(q[i] * scale[f] + shift[f]);
}

// ---------------------------------------------------------------- fused sim (WMMA) + two-level top-k + softmax
__global__ __launch_bounds__(256) void k_sim_topk(const _Float16* __restrict__ qn,
                                                  const _Float16* __restrict__ kf,
                                                  float* __restrict__ wts,
                                                  int* __restrict__ inds) {
    __shared__ _Float16 As[2 * 16 * 128];             // 8KB: [p][tok][d]
    __shared__ float simL[2][16][NKEY];               // 32KB
    __shared__ float tv[2][16][8];
    __shared__ int   ti[2][16][8];

    int tid = threadIdx.x, lane = tid & 31, wave = tid >> 5;
    int t0 = blockIdx.x * 16, h = (int)blockIdx.y;

    { // stage normalized q slices for both planes (coalesced dwords)
        const unsigned int* src = (const unsigned int*)qn;
        unsigned int* dst = (unsigned int*)As;
        for (int li = tid; li < 2048; li += 256) {
            int p = li >> 10, row = (li >> 6) & 15, cu = li & 63;
            dst[li] = src[(t0 + row) * 1024 + p * 512 + h * 64 + cu];
        }
    }
    __syncthreads();

    { // wave w: plane = w>>2, key-tiles {sw, sw+4, sw+8, sw+12}
        int p = wave >> 2, sw = wave & 3;
        int rowA  = lane & 15;
        int kOffA = (lane & 16);
        const _Float16* Ap = As + (p * 16 + rowA) * 128;
        v8f acc[4];
        #pragma unroll
        for (int j = 0; j < 4; ++j)
            #pragma unroll
            for (int r = 0; r < 8; ++r) acc[j][r] = 0.0f;

        const v16h* bbase = (const v16h*)kf + ((p * 8 + h) * 64) * 32 + lane;
        for (int ks = 0; ks < 4; ++ks) {
            v16h a;
            const _Float16* ar = Ap + ks * 32;
            #pragma unroll
            for (int i = 0; i < 8; ++i) a[i] = ar[kOffA + i];
            #pragma unroll
            for (int i = 0; i < 8; ++i) a[8 + i] = ar[kOffA + 8 + i];
            #pragma unroll
            for (int j = 0; j < 4; ++j) {
                int nt = sw + 4 * j;
                v16h b = bbase[(nt * 4 + ks) * 32];
                acc[j] = __builtin_amdgcn_wmma_f32_16x16x32_f16(false, a, false, b, (short)0,
                                                                acc[j], false, false);
            }
        }
        int rBase = (lane & 16) >> 1, cl = lane & 15;
        #pragma unroll
        for (int j = 0; j < 4; ++j) {
            int nt = sw + 4 * j;
            #pragma unroll
            for (int r = 0; r < 8; ++r)
                simL[p][rBase + r][nt * 16 + cl] = acc[j][r];
        }
    }
    __syncthreads();

    // per-plane top-8 of 256 (stable: earlier index wins on ties, like lax.top_k)
    if (tid < 32) {
        int p = tid >> 4, t = tid & 15;
        float bv[8]; int bi[8];
        #pragma unroll
        for (int k = 0; k < 8; ++k) { bv[k] = -3.4e38f; bi[k] = 0; }
        for (int k = 0; k < NKEY; ++k) top8_insert(bv, bi, simL[p][t][k], k);
        #pragma unroll
        for (int k = 0; k < 8; ++k) { tv[p][t][k] = bv[k]; ti[p][t][k] = bi[k]; }
    }
    __syncthreads();

    // cross-product top-8 of 64 + softmax
    if (tid < 16) {
        int t = tid;
        float bv[8]; int bi[8];
        #pragma unroll
        for (int k = 0; k < 8; ++k) { bv[k] = -3.4e38f; bi[k] = 0; }
        for (int i = 0; i < 8; ++i)
            for (int j = 0; j < 8; ++j)
                top8_insert(bv, bi, tv[0][t][i] + tv[1][t][j], ti[0][t][i] * NKEY + ti[1][t][j]);
        float m = bv[0], s = 0.f, e[8];
        #pragma unroll
        for (int k = 0; k < 8; ++k) { e[k] = __expf(bv[k] - m); s += e[k]; }
        float inv = 1.0f / s;
        int base = ((t0 + t) * NH + h) * TKP;
        #pragma unroll
        for (int k = 0; k < 8; ++k) { wts[base + k] = e[k] * inv; inds[base + k] = bi[k]; }
    }
}

// ---------------------------------------------------------------- expert gather-compute (HBM-bound stage)
// ~1.07 GB of data-dependent row gathers dominates the whole op (~46us @ 23.3 TB/s).
// Indices are known ahead of use, so issue global_prefetch (locality=3 so the
// AMDGPU backend keeps it) for upcoming expert rows while the current row's FMAs
// drain. Prefetch indices are clamped with a select (no branch) so the loops
// stay unrollable; tail prefetches of resident rows are harmless.
__global__ __launch_bounds__(256) void k_expert(const float* __restrict__ x,
                                                const float* __restrict__ down,
                                                const float* __restrict__ up,
                                                const float* __restrict__ wts,
                                                const int* __restrict__ inds,
                                                float* __restrict__ out) {
    __shared__ float xs[NDIM];
    __shared__ float coeff[64];
    __shared__ int   eidx[64];
    int tid = threadIdx.x, lane = tid & 31, wave = tid >> 5;
    int t = blockIdx.x;

    { // stage token activation, 4KB
        const float4* src = (const float4*)(x + (size_t)t * NDIM);
        ((float4*)xs)[tid] = src[tid];
    }
    if (tid < 64) eidx[tid] = inds[(size_t)t * 64 + tid];
    __syncthreads();

    // warm the gather stream: wave-wide prefetch of this wave's first down row
    // (lane*128B stride covers the full 4KB row in one shot)
    __builtin_prefetch(down + (size_t)eidx[wave * 8] * NDIM + lane * 32, 0, 3);

    // phase A: each wave owns 8 experts; 512B coalesced reads, wave32 butterfly reduce
    for (int ei = 0; ei < 8; ++ei) {
        int e  = wave * 8 + ei;
        int en = (ei < 7) ? (e + 1) : e;   // branch-free next-row prefetch index
        __builtin_prefetch(down + (size_t)eidx[en] * NDIM + lane * 32, 0, 3);
        const float* dr = down + (size_t)eidx[e] * NDIM;
        float partial = 0.f;
        #pragma unroll
        for (int cch = 0; cch < 8; ++cch) {
            int d = cch * 128 + lane * 4;
            float4 a = *(const float4*)(xs + d);
            float4 b = *(const float4*)(dr + d);
            partial += a.x * b.x + a.y * b.y + a.z * b.z + a.w * b.w;
        }
        for (int off = 16; off > 0; off >>= 1) partial += __shfl_xor(partial, off, 32);
        if (lane == 0) coeff[e] = wts[(size_t)t * 64 + e] * gelu_new_f(partial);
    }
    __syncthreads();

    // phase B: out[t, d0..d0+3] = sum_e coeff[e] * up[idx_e, d0..]; reads fully coalesced per expert
    int d0 = tid * 4;
    // prime two up-rows of prefetch ahead of the accumulation loop
    __builtin_prefetch(up + (size_t)eidx[0] * NDIM + d0, 0, 3);
    __builtin_prefetch(up + (size_t)eidx[1] * NDIM + d0, 0, 3);
    float ax = 0.f, ay = 0.f, az = 0.f, aw = 0.f;
    for (int e = 0; e < 64; ++e) {
        int ep = (e + 2 < 64) ? (e + 2) : 63;   // branch-free lookahead
        __builtin_prefetch(up + (size_t)eidx[ep] * NDIM + d0, 0, 3);
        float cf = coeff[e];
        float4 u = *(const float4*)(up + (size_t)eidx[e] * NDIM + d0);
        ax += cf * u.x; ay += cf * u.y; az += cf * u.z; aw += cf * u.w;
    }
    float4 o; o.x = ax; o.y = ay; o.z = az; o.w = aw;
    *(float4*)(out + (size_t)t * NDIM + d0) = o;
}

// ---------------------------------------------------------------- launcher
extern "C" void kernel_launch(void* const* d_in, const int* in_sizes, int n_in,
                              void* d_out, int out_size, void* d_ws, size_t ws_size,
                              hipStream_t stream) {
    (void)in_sizes; (void)n_in; (void)out_size; (void)ws_size;
    const float* x     = (const float*)d_in[0];
    const float* w_q   = (const float*)d_in[1];
    const float* gamma = (const float*)d_in[2];
    const float* beta  = (const float*)d_in[3];
    const float* keys  = (const float*)d_in[4];
    const float* down  = (const float*)d_in[5];
    const float* up    = (const float*)d_in[6];
    float* out = (float*)d_out;

    char* ws = (char*)d_ws;
    size_t off = 0;
    auto alloc = [&](size_t bytes) { size_t o = off; off += (bytes + 255) & ~(size_t)255; return o; };
    float*    q     = (float*)(ws + alloc((size_t)T_TOK * NFEAT * 4));     // 16 MB
    _Float16* xh    = (_Float16*)(ws + alloc((size_t)T_TOK * NDIM * 2));   // 4 MB
    _Float16* wf    = (_Float16*)(ws + alloc((size_t)NDIM * NFEAT * 2));   // 4 MB
    _Float16* kf    = (_Float16*)(ws + alloc((size_t)524288 * 2));         // 1 MB
    _Float16* qn    = (_Float16*)(ws + alloc((size_t)T_TOK * NFEAT * 2));  // 8 MB
    float*    scale = (float*)(ws + alloc(NFEAT * 4));
    float*    shift = (float*)(ws + alloc(NFEAT * 4));
    float*    wtsb  = (float*)(ws + alloc((size_t)T_TOK * 64 * 4));
    int*      indb  = (int*)(ws + alloc((size_t)T_TOK * 64 * 4));

    k_cvt_x    <<<8192, 256, 0, stream>>>(x, xh, T_TOK * NDIM);
    k_cvt_wfrag<<<8192, 256, 0, stream>>>(w_q, wf);
    k_cvt_keysf<<<2048, 256, 0, stream>>>(keys, kf);
    k_gemm_q   <<<dim3(16, 128), 256, 0, stream>>>(xh, wf, q);
    k_bn       <<<8, 256, 0, stream>>>(q, gamma, beta, scale, shift);
    k_qn       <<<16384, 256, 0, stream>>>(q, scale, shift, qn);
    k_sim_topk <<<dim3(128, 8), 256, 0, stream>>>(qn, kf, wtsb, indb);
    k_expert   <<<2048, 256, 0, stream>>>(x, down, up, wtsb, indb, out);
}
